// SparseViT5SlideEncoder_79577154060939
// MI455X (gfx1250) — matrix operand order repaired and checked by reading, not compiled, not run on hardware
//
#include <hip/hip_runtime.h>
#include <hip/hip_bf16.h>
#include <math.h>

// ---------------- model constants ----------------
constexpr int B_   = 2;
constexpr int N_   = 4096;
constexpr int C_   = 2;          // NUM_CLS
constexpr int S_   = N_ + C_;    // 4098
constexpr int D_   = 384;        // EMBED
constexpr int H_   = 6;          // HEADS
constexpr int DH_  = 64;         // head dim
constexpr int L_   = 6;          // DEPTH
constexpr int DFF_ = 1536;
constexpr int QKV_ = 3 * D_;     // 1152
constexpr int WIN_ = 4;
constexpr int DIL_ = 2;
constexpr int NKEY_ = C_ + (2 * WIN_ + 1);   // 11 keys per patch token

// ---------------- workspace layout (floats) ----------------
constexpr size_t OFF_H    = 0;
constexpr size_t SZ_H     = (size_t)B_ * S_ * D_;
constexpr size_t OFF_NORM = OFF_H + SZ_H;
constexpr size_t OFF_ATTN = OFF_NORM + SZ_H;
constexpr size_t OFF_BIG  = OFF_ATTN + SZ_H;               // qkv / mlp-mid shared
constexpr size_t SZ_BIG   = (size_t)B_ * S_ * DFF_;
constexpr size_t OFF_CLS  = OFF_BIG + SZ_BIG;              // 24 * 4098 scores

// ---------------- WMMA types ----------------
typedef __attribute__((ext_vector_type(16))) __bf16 v16bf;
typedef __attribute__((ext_vector_type(8)))  float  v8f;

union BFrag { int i[8]; v16bf v; };

__device__ __forceinline__ unsigned f2bf_hi(float f) {
    unsigned u = __float_as_uint(f);
    return (u + 0x7FFFu + ((u >> 16) & 1u)) >> 16;   // round-to-nearest-even
}
__device__ __forceinline__ unsigned pk_bf2(float lo, float hi) {
    return f2bf_hi(lo) | (f2bf_hi(hi) << 16);
}

// =======================================================================
// Tiled bf16-WMMA GEMM:  C[M,N] = A[M,K] (f32, ldA) * W[N,K]^T (f32 row-major)
// 128x128 tile / block of 256 threads (8 waves); wave computes 32x64.
// N must be a multiple of 128 (all call sites: 384/1152/1536).
// mode 0: C = acc            (no bias)
// mode 1: C = acc + bias
// mode 2: C = gelu(acc + bias)           (exact erf gelu)
// mode 3: C += gamma * (acc + bias)      (residual)
// Output row remap: out_row = (r/rpb_in)*rpb_out + row_off + r%rpb_in
// =======================================================================
__global__ __launch_bounds__(256)
void gemm_bf16_wmma(const float* __restrict__ A, const float* __restrict__ W,
                    const float* __restrict__ bias, const float* __restrict__ gamma,
                    float* __restrict__ Cout,
                    int M, int K, int N, int ldA, int ldC, int mode,
                    int rpb_in, int rpb_out, int row_off)
{
    // bf16 pairs packed in dwords; row stride 20 dwords (80B -> 16B-aligned rows)
    __shared__ __align__(16) unsigned lA[128][20];
    __shared__ __align__(16) unsigned lB[128][20];

    const int t    = threadIdx.x;
    const int lane = t & 31;
    const int wave = t >> 5;
    const int wm   = wave & 3;          // 4 wave-rows of 32
    const int wn   = wave >> 2;         // 2 wave-cols of 64
    const int mBase = wm * 32;
    const int nBase = wn * 64;
    const int m0 = blockIdx.x * 128;
    const int n0 = blockIdx.y * 128;

    const int mlane = lane & 15;
    const int hh    = lane >> 4;

    v8f acc[2][4];
#pragma unroll
    for (int i = 0; i < 2; i++)
#pragma unroll
        for (int j = 0; j < 4; j++) acc[i][j] = (v8f){};

    const int lrow = t >> 1;             // 0..127
    const int lcol = (t & 1) << 4;       // 0 or 16 (float index within 32-wide tile)
    const bool fullM = (m0 + 128) <= M;  // block-uniform: hot path has no guards

    const float* aRowP = A + (size_t)(m0 + lrow) * ldA + lcol;
    const float* wRowP = W + (size_t)(n0 + lrow) * K + lcol;
    unsigned* dstA = &lA[lrow][lcol >> 1];
    unsigned* dstB = &lB[lrow][lcol >> 1];

    for (int kt = 0; kt < K; kt += 32) {
        // ---- stage W tile (128 x 32): unguarded (N multiple of 128) ----
        {
            const float4* wp = (const float4*)(wRowP + kt);
            unsigned p[8];
#pragma unroll
            for (int i = 0; i < 4; i++) {
                float4 v = wp[i];
                p[2 * i]     = pk_bf2(v.x, v.y);
                p[2 * i + 1] = pk_bf2(v.z, v.w);
            }
            ((uint4*)dstB)[0] = make_uint4(p[0], p[1], p[2], p[3]);
            ((uint4*)dstB)[1] = make_uint4(p[4], p[5], p[6], p[7]);
            if (kt + 32 < K) __builtin_prefetch(wRowP + kt + 32, 0, 1);
        }
        // ---- stage A tile (128 x 32) ----
        if (fullM) {
            const float4* ap = (const float4*)(aRowP + kt);
            unsigned p[8];
#pragma unroll
            for (int i = 0; i < 4; i++) {
                float4 v = ap[i];
                p[2 * i]     = pk_bf2(v.x, v.y);
                p[2 * i + 1] = pk_bf2(v.z, v.w);
            }
            ((uint4*)dstA)[0] = make_uint4(p[0], p[1], p[2], p[3]);
            ((uint4*)dstA)[1] = make_uint4(p[4], p[5], p[6], p[7]);
        } else {
            const bool av = (m0 + lrow) < M;
            unsigned p[8];
#pragma unroll
            for (int i = 0; i < 8; i++) {
                float a0 = av ? aRowP[kt + 2 * i]     : 0.0f;
                float a1 = av ? aRowP[kt + 2 * i + 1] : 0.0f;
                p[i] = pk_bf2(a0, a1);
            }
            ((uint4*)dstA)[0] = make_uint4(p[0], p[1], p[2], p[3]);
            ((uint4*)dstA)[1] = make_uint4(p[4], p[5], p[6], p[7]);
        }
        __syncthreads();

        // ---- build fragments per CDNA5 16-bit layouts (dword = bf16 pair) ----
        BFrag af[2];
#pragma unroll
        for (int mt = 0; mt < 2; mt++) {
            const int arow = mBase + mt * 16 + mlane;
#pragma unroll
            for (int v = 0; v < 8; v++) {
                const int di = (v < 4) ? (hh * 4 + v) : (8 + hh * 4 + (v - 4));
                af[mt].i[v] = (int)lA[arow][di];
            }
        }
        BFrag bfr[4];
#pragma unroll
        for (int nt = 0; nt < 4; nt++) {
            const int brow = nBase + nt * 16 + mlane;
#pragma unroll
            for (int v = 0; v < 8; v++) {
                bfr[nt].i[v] = (int)lB[brow][(v < 4) ? (hh * 4 + v) : (8 + hh * 4 + (v - 4))];
            }
        }

        // ---- 8 WMMAs: D = A*B + C, f32 accumulate ----
#pragma unroll
        for (int mt = 0; mt < 2; mt++)
#pragma unroll
            for (int nt = 0; nt < 4; nt++)
                acc[mt][nt] = __builtin_amdgcn_wmma_f32_16x16x32_bf16(
                    false, af[mt].v, false, bfr[nt].v,
                    (short)0, acc[mt][nt], false, false);

        __syncthreads();
    }

    // ---- epilogue (16x16 f32 C layout: col = lane%16, row = v + 8*(lane/16)) ----
#pragma unroll
    for (int mt = 0; mt < 2; mt++) {
#pragma unroll
        for (int nt = 0; nt < 4; nt++) {
            const int col = n0 + nBase + nt * 16 + mlane;
#pragma unroll
            for (int v = 0; v < 8; v++) {
                const int r = m0 + mBase + mt * 16 + v + 8 * hh;
                if (r >= M) continue;
                const int orow = (r / rpb_in) * rpb_out + row_off + (r % rpb_in);
                float x = acc[mt][nt][v];
                float* o = Cout + (size_t)orow * ldC + col;
                if (mode == 0) {
                    *o = x;
                } else if (mode == 1) {
                    *o = x + bias[col];
                } else if (mode == 2) {
                    float y = x + bias[col];
                    *o = 0.5f * y * (1.0f + erff(y * 0.70710678118654752f));
                } else {
                    float y = x + bias[col];
                    *o += gamma[col] * y;
                }
            }
        }
    }
}

// =======================================================================
// RMSNorm: y[row] = x[row] * rsqrt(mean(x^2)+eps) * w  (one 128-thread block/row)
// =======================================================================
__global__ __launch_bounds__(128)
void rmsnorm_k(const float* __restrict__ x, const float* __restrict__ w,
               float* __restrict__ y)
{
    __shared__ float red[128];
    const int row = blockIdx.x;
    const int t = threadIdx.x;
    const float* xr = x + (size_t)row * D_;
    float p = 0.0f;
    for (int d = t; d < D_; d += 128) { float v = xr[d]; p += v * v; }
    red[t] = p; __syncthreads();
    for (int s = 64; s > 0; s >>= 1) { if (t < s) red[t] += red[t + s]; __syncthreads(); }
    const float inv = rsqrtf(red[0] / (float)D_ + 1e-6f);
    float* yr = y + (size_t)row * D_;
    for (int d = t; d < D_; d += 128) yr[d] = xr[d] * inv * w[d];
}

// =======================================================================
// cls token init: h[b, c, :] = cls_tokens[c, :]
// =======================================================================
__global__ void cls_init_k(const float* __restrict__ cls, float* __restrict__ h)
{
    int tid = blockIdx.x * blockDim.x + threadIdx.x;
    if (tid >= B_ * C_ * D_) return;
    int d = tid % D_;
    int c = (tid / D_) % C_;
    int b = tid / (D_ * C_);
    h[((size_t)(b * S_ + c)) * D_ + d] = cls[c * D_ + d];
}

// =======================================================================
// RoPE (in-place) on q,k of patch rows of qkv buffer.
// thread: (b, n, h, j) with j in 0..31 owning dims j and j+32.
// =======================================================================
__global__ void rope_k(float* __restrict__ qkv, const int* __restrict__ coords)
{
    int tid = blockIdx.x * blockDim.x + threadIdx.x;
    if (tid >= B_ * N_ * H_ * 32) return;
    int j = tid & 31;
    int h = (tid >> 5) % H_;
    int n = (tid / (32 * H_)) % N_;
    int b = tid / (32 * H_ * N_);
    int cidx = j >> 4;          // coordinate 0 or 1
    int f    = j & 15;          // frequency index
    float pos = (float)coords[((size_t)(b * N_ + n)) * 2 + cidx] * 1e-5f;
    float inv = __powf(10000.0f, -(float)f / 16.0f);
    float ang = pos * inv;
    float cs = __cosf(ang), sn = __sinf(ang);
    float* row = qkv + ((size_t)(b * S_ + C_ + n)) * QKV_ + h * DH_;
#pragma unroll
    for (int comp = 0; comp < 2; comp++) {       // q (0), k (D_)
        float* p = row + comp * D_;
        float t1 = p[j], t2 = p[j + 32];
        p[j]      = t1 * cs - t2 * sn;
        p[j + 32] = t1 * sn + t2 * cs;
    }
}

// =======================================================================
// Windowed sparse attention for patch tokens: 2 cls keys + 9 dilated window keys.
// one thread per (b, n, h)
// =======================================================================
__global__ void win_attn_k(const float* __restrict__ qkv, float* __restrict__ ao)
{
    int tid = blockIdx.x * blockDim.x + threadIdx.x;
    if (tid >= B_ * N_ * H_) return;
    int h = tid % H_;
    int n = (tid / H_) % N_;
    int b = tid / (H_ * N_);
    const float* base = qkv + (size_t)b * S_ * QKV_;
    const float* q = base + ((size_t)(C_ + n)) * QKV_ + h * DH_;

    float sc[NKEY_];
    int   idxs[2 * WIN_ + 1];
#pragma unroll
    for (int c = 0; c < C_; c++) {
        const float* k = base + (size_t)c * QKV_ + D_ + h * DH_;
        float d = 0.0f;
#pragma unroll
        for (int e = 0; e < DH_; e++) d += q[e] * k[e];
        sc[c] = d * 0.125f;
    }
#pragma unroll
    for (int w = 0; w < 2 * WIN_ + 1; w++) {
        int idx = n + DIL_ * (w - WIN_);
        idxs[w] = idx;
        if (idx >= 0 && idx < N_) {
            const float* k = base + ((size_t)(C_ + idx)) * QKV_ + D_ + h * DH_;
            float d = 0.0f;
#pragma unroll
            for (int e = 0; e < DH_; e++) d += q[e] * k[e];
            sc[C_ + w] = d * 0.125f;
        } else {
            sc[C_ + w] = -1e30f;
        }
    }
    float m = sc[0];
#pragma unroll
    for (int j = 1; j < NKEY_; j++) m = fmaxf(m, sc[j]);
    float z = 0.0f;
#pragma unroll
    for (int j = 0; j < NKEY_; j++) { sc[j] = __expf(sc[j] - m); z += sc[j]; }
    float iz = 1.0f / z;
#pragma unroll
    for (int j = 0; j < NKEY_; j++) sc[j] *= iz;

    float* o = ao + ((size_t)(b * S_ + C_ + n)) * D_ + h * DH_;
    for (int e = 0; e < DH_; e++) {
        float a = sc[0] * base[2 * D_ + h * DH_ + e]
                + sc[1] * base[(size_t)QKV_ + 2 * D_ + h * DH_ + e];
#pragma unroll
        for (int w = 0; w < 2 * WIN_ + 1; w++) {
            int idx = idxs[w];
            if ((unsigned)idx < (unsigned)N_)
                a += sc[C_ + w] * base[((size_t)(C_ + idx)) * QKV_ + 2 * D_ + h * DH_ + e];
        }
        o[e] = a;
    }
}

// =======================================================================
// cls-query attention over all S keys: scores -> softmax -> weighted V
// =======================================================================
__global__ void cls_scores_k(const float* __restrict__ qkv, float* __restrict__ cs)
{
    int tid = blockIdx.x * blockDim.x + threadIdx.x;
    if (tid >= B_ * C_ * H_ * S_) return;
    int s = tid % S_;
    int h = (tid / S_) % H_;
    int c = (tid / (S_ * H_)) % C_;
    int b = tid / (S_ * H_ * C_);
    const float* base = qkv + (size_t)b * S_ * QKV_;
    const float* q = base + (size_t)c * QKV_ + h * DH_;          // un-roped cls q
    const float* k = base + (size_t)s * QKV_ + D_ + h * DH_;
    float d = 0.0f;
#pragma unroll
    for (int e = 0; e < DH_; e++) d += q[e] * k[e];
    cs[(((size_t)((b * C_ + c) * H_ + h)) * S_) + s] = d * 0.125f;
}

__global__ __launch_bounds__(256)
void cls_softmax_k(float* __restrict__ cs)
{
    __shared__ float red[256];
    __shared__ float mz[2];
    const int t = threadIdx.x;
    float* p = cs + (size_t)blockIdx.x * S_;

    float lm = -3.4e38f;
    for (int s = t; s < S_; s += 256) lm = fmaxf(lm, p[s]);
    red[t] = lm; __syncthreads();
    for (int s = 128; s > 0; s >>= 1) { if (t < s) red[t] = fmaxf(red[t], red[t + s]); __syncthreads(); }
    if (t == 0) mz[0] = red[0];
    __syncthreads();
    const float m = mz[0];

    float lz = 0.0f;
    for (int s = t; s < S_; s += 256) { float e = __expf(p[s] - m); p[s] = e; lz += e; }
    __syncthreads();
    red[t] = lz; __syncthreads();
    for (int s = 128; s > 0; s >>= 1) { if (t < s) red[t] += red[t + s]; __syncthreads(); }
    if (t == 0) mz[1] = red[0];
    __syncthreads();
    const float iz = 1.0f / mz[1];
    for (int s = t; s < S_; s += 256) p[s] *= iz;
}

__global__ void cls_out_k(const float* __restrict__ cs, const float* __restrict__ qkv,
                          float* __restrict__ ao)
{
    int tid = blockIdx.x * blockDim.x + threadIdx.x;
    if (tid >= B_ * C_ * H_ * DH_) return;
    int d = tid % DH_;
    int h = (tid / DH_) % H_;
    int c = (tid / (DH_ * H_)) % C_;
    int b = tid / (DH_ * H_ * C_);
    const float* p = cs + ((size_t)((b * C_ + c) * H_ + h)) * S_;
    const float* base = qkv + (size_t)b * S_ * QKV_ + 2 * D_ + h * DH_ + d;
    float a = 0.0f;
    for (int s = 0; s < S_; s++) a += p[s] * base[(size_t)s * QKV_];
    ao[((size_t)(b * S_ + c)) * D_ + h * DH_ + d] = a;
}

// =======================================================================
// Final pooling + head: one 384-thread block per batch element
// =======================================================================
__global__ __launch_bounds__(384)
void pool_head_k(const float* __restrict__ hn, const float* __restrict__ wp,
                 const float* __restrict__ wh, const float* __restrict__ bh,
                 float* __restrict__ out)
{
    __shared__ float r0[384], r1[384];
    __shared__ float ww[2];
    const int b = blockIdx.x, t = threadIdx.x;
    const float* c0 = hn + (size_t)(b * S_) * D_;
    const float* c1 = c0 + D_;
    r0[t] = c0[t] * wp[t];
    r1[t] = c1[t] * wp[t];
    __syncthreads();
    for (int s = 256; s > 0; s >>= 1) {
        if (t < s && t + s < 384) { r0[t] += r0[t + s]; r1[t] += r1[t + s]; }
        __syncthreads();
    }
    if (t == 0) {
        float s0 = r0[0], s1 = r1[0];
        float m = fmaxf(s0, s1);
        float e0 = __expf(s0 - m), e1 = __expf(s1 - m);
        float z = e0 + e1;
        ww[0] = e0 / z; ww[1] = e1 / z;
    }
    __syncthreads();
    float pooled = ww[0] * c0[t] + ww[1] * c1[t];
    __syncthreads();
    r0[t] = pooled * wh[t];
    __syncthreads();
    for (int s = 256; s > 0; s >>= 1) {
        if (t < s && t + s < 384) r0[t] += r0[t + s];
        __syncthreads();
    }
    if (t == 0) out[b] = r0[0] + bh[0];
}

// =======================================================================
// host launcher
// =======================================================================
extern "C" void kernel_launch(void* const* d_in, const int* in_sizes, int n_in,
                              void* d_out, int out_size, void* d_ws, size_t ws_size,
                              hipStream_t stream)
{
    const float* x       = (const float*)d_in[0];
    const int*   coords  = (const int*)  d_in[1];
    const float* w_in    = (const float*)d_in[2];
    const float* b_in    = (const float*)d_in[3];
    const float* cls_tok = (const float*)d_in[4];
    const float* w_qkv   = (const float*)d_in[5];
    const float* w_proj  = (const float*)d_in[6];
    const float* b_proj  = (const float*)d_in[7];
    const float* n1      = (const float*)d_in[8];
    const float* n2      = (const float*)d_in[9];
    const float* g1      = (const float*)d_in[10];
    const float* g2      = (const float*)d_in[11];
    const float* w_fc1   = (const float*)d_in[12];
    const float* b_fc1   = (const float*)d_in[13];
    const float* w_fc2   = (const float*)d_in[14];
    const float* b_fc2   = (const float*)d_in[15];
    const float* n_f     = (const float*)d_in[16];
    const float* w_pool  = (const float*)d_in[17];
    const float* w_head  = (const float*)d_in[18];
    const float* b_head  = (const float*)d_in[19];
    float* out = (float*)d_out;

    float* ws   = (float*)d_ws;
    float* h    = ws + OFF_H;
    float* norm = ws + OFF_NORM;
    float* attn = ws + OFF_ATTN;
    float* big  = ws + OFF_BIG;     // qkv then mlp-mid (reused)
    float* csb  = ws + OFF_CLS;

    const int Mrows = B_ * S_;                       // 8196
    const int MB    = (Mrows + 127) / 128;           // 65
    const dim3 blk(256);

    // h[:, C:, :] = x @ w_in.T + b_in   (row remap 4096 -> 4098 with +2 offset)
    gemm_bf16_wmma<<<dim3((B_ * N_) / 128, D_ / 128), blk, 0, stream>>>(
        x, w_in, b_in, nullptr, h, B_ * N_, 1280, D_, 1280, D_, 1, N_, S_, C_);
    cls_init_k<<<(B_ * C_ * D_ + 255) / 256, blk, 0, stream>>>(cls_tok, h);

    for (int i = 0; i < L_; i++) {
        // attn branch
        rmsnorm_k<<<Mrows, 128, 0, stream>>>(h, n1 + (size_t)i * D_, norm);
        gemm_bf16_wmma<<<dim3(MB, QKV_ / 128), blk, 0, stream>>>(
            norm, w_qkv + (size_t)i * QKV_ * D_, nullptr, nullptr, big,
            Mrows, D_, QKV_, D_, QKV_, 0, S_, S_, 0);
        rope_k<<<(B_ * N_ * H_ * 32 + 255) / 256, blk, 0, stream>>>(big, coords);
        win_attn_k<<<(B_ * N_ * H_ + 255) / 256, blk, 0, stream>>>(big, attn);
        cls_scores_k<<<(B_ * C_ * H_ * S_ + 255) / 256, blk, 0, stream>>>(big, csb);
        cls_softmax_k<<<B_ * C_ * H_, blk, 0, stream>>>(csb);
        cls_out_k<<<(B_ * C_ * H_ * DH_ + 255) / 256, blk, 0, stream>>>(csb, big, attn);
        gemm_bf16_wmma<<<dim3(MB, D_ / 128), blk, 0, stream>>>(
            attn, w_proj + (size_t)i * D_ * D_, b_proj + (size_t)i * D_,
            g1 + (size_t)i * D_, h, Mrows, D_, D_, D_, D_, 3, S_, S_, 0);

        // mlp branch
        rmsnorm_k<<<Mrows, 128, 0, stream>>>(h, n2 + (size_t)i * D_, norm);
        gemm_bf16_wmma<<<dim3(MB, DFF_ / 128), blk, 0, stream>>>(
            norm, w_fc1 + (size_t)i * DFF_ * D_, b_fc1 + (size_t)i * DFF_,
            nullptr, big, Mrows, D_, DFF_, D_, DFF_, 2, S_, S_, 0);
        gemm_bf16_wmma<<<dim3(MB, D_ / 128), blk, 0, stream>>>(
            big, w_fc2 + (size_t)i * D_ * DFF_, b_fc2 + (size_t)i * D_,
            g2 + (size_t)i * D_, h, Mrows, DFF_, D_, DFF_, D_, 3, S_, S_, 0);
    }

    rmsnorm_k<<<Mrows, 128, 0, stream>>>(h, n_f, norm);
    pool_head_k<<<B_, 384, 0, stream>>>(norm, w_pool, w_head, b_head, out);
}
